// DecoderWithAttention_choice_300647710801
// MI455X (gfx1250) — matrix operand order repaired and checked by reading, not compiled, not run on hardware
//
#include <hip/hip_runtime.h>
#include <hip/hip_bf16.h>
#include <stdint.h>

// ---------------------------------------------------------------------------
// Problem constants (from reference): B=64, T=52, V=10000, E=512, H=512, DE=2048
// ---------------------------------------------------------------------------
#define NB    64
#define NT    52
#define TD    51          // T-1 decode steps
#define NV    10000
#define NE    512
#define NH    512
#define NDE   2048
#define G4H   2048        // 4*H gate width

typedef __attribute__((ext_vector_type(16))) __bf16 v16bf;
typedef __attribute__((ext_vector_type(8)))  __bf16 v8bf;
typedef __attribute__((ext_vector_type(8)))  float  v8f;

__device__ __forceinline__ float sigmoidf_(float x) { return 1.0f / (1.0f + __expf(-x)); }

// A-fragment: 16x32 bf16, row-major source (lda elements per row).
// lanes 0-15: M=lane, elems 0-7 -> K=k0..k0+7, elems 8-15 -> K=k0+16..k0+23
// lanes16-31: M=lane-16, elems 0-7 -> K=k0+8.., elems 8-15 -> K=k0+24..
__device__ __forceinline__ v16bf load_a_frag(const __bf16* __restrict__ A, int lda,
                                             int row0, int k0, int lane) {
  int row = row0 + (lane & 15);
  int hi  = (lane >> 4) & 1;
  const __bf16* p = A + (size_t)row * lda + k0 + hi * 8;
  v8bf lo = *(const v8bf*)p;
  v8bf hh = *(const v8bf*)(p + 16);
  return __builtin_shufflevector(lo, hh, 0,1,2,3,4,5,6,7,8,9,10,11,12,13,14,15);
}

// B-fragment: 32x16 bf16, computing A @ W^T with W row-major (N x K, ldw).
// B(k,n) = W[n][k]; lanes 0-15: N=lane, elems 0-7 -> K=k0..k0+7, elems 8-15 -> K=k0+16..k0+23;
// lanes 16-31: elems 0-7 -> K=k0+8.., elems 8-15 -> K=k0+24...
__device__ __forceinline__ v16bf load_b_frag(const __bf16* __restrict__ W, int ldw,
                                             int n0, int k0, int lane) {
  int col = n0 + (lane & 15);
  int hi  = (lane >> 4) & 1;
  const __bf16* p = W + (size_t)col * ldw + k0 + hi * 8;
  v8bf lo = *(const v8bf*)p;
  v8bf hh = *(const v8bf*)(p + 16);
  return __builtin_shufflevector(lo, hh, 0,1,2,3,4,5,6,7,8,9,10,11,12,13,14,15);
}

__device__ __forceinline__ v8f wmma_bf16(v16bf a, v16bf b, v8f c) {
  return __builtin_amdgcn_wmma_f32_16x16x32_bf16(false, a, false, b, (short)0, c,
                                                 false, false);
}

// ---------------------------------------------------------------------------
// prep: lens, stable argsort (descending), sorted caps, reversed caps,
// plus int-valued outputs written as floats into d_out tail.
// ---------------------------------------------------------------------------
__global__ void __launch_bounds__(64)
prep_kernel(const int* __restrict__ cap_len,   // (64,1)
            const int* __restrict__ caps_in,   // (64,52)
            int* __restrict__ lens_s, int* __restrict__ order_i,
            int* __restrict__ caps_s, int* __restrict__ caps_rev,
            float* __restrict__ out_caps, float* __restrict__ out_caps_rev,
            float* __restrict__ out_lens, float* __restrict__ out_order) {
  __shared__ int slen[NB], sord[NB];
  int tid = threadIdx.x;
  if (tid == 0) {
    int L[NB], O[NB];
    for (int i = 0; i < NB; i++) { L[i] = cap_len[i]; O[i] = i; }
    for (int i = 1; i < NB; i++) {           // stable insertion sort, descending
      int kl = L[i], ko = O[i], j = i - 1;
      while (j >= 0 && L[j] < kl) { L[j+1] = L[j]; O[j+1] = O[j]; j--; }
      L[j+1] = kl; O[j+1] = ko;
    }
    for (int i = 0; i < NB; i++) { slen[i] = L[i]; sord[i] = O[i]; }
  }
  __syncthreads();
  int b = tid, lb = slen[b], ob = sord[b];
  lens_s[b] = lb; order_i[b] = ob;
  out_lens[b] = (float)(lb - 1); out_order[b] = (float)ob;
  for (int t = 0; t < NT; t++) {
    int c = caps_in[ob * NT + t];
    caps_s[b * NT + t] = c; out_caps[b * NT + t] = (float)c;
  }
  for (int t = 0; t < NT; t++) {
    bool valid = t < lb;
    int ridx = valid ? (lb - 1 - t) : 0;
    int cr = valid ? caps_s[b * NT + ridx] : 0;
    caps_rev[b * NT + t] = cr; out_caps_rev[b * NT + t] = (float)cr;
  }
}

// gather encoder rows by sort order, convert to bf16
__global__ void __launch_bounds__(256)
enc_gather_kernel(const float* __restrict__ enc, const int* __restrict__ order,
                  __bf16* __restrict__ dst) {
  int i = blockIdx.x * 256 + threadIdx.x;      // 64*2048
  if (i >= NB * NDE) return;
  int k = i & (NDE - 1), b = i >> 11;
  dst[i] = (__bf16)enc[(size_t)order[b] * NDE + k];
}

// embedding gather -> bf16, layout [t][b][E]
__global__ void __launch_bounds__(256)
embed_kernel(const float* __restrict__ table, const int* __restrict__ caps,
             __bf16* __restrict__ dst) {
  int i = blockIdx.x * 256 + threadIdx.x;      // TD*64*512
  if (i >= TD * NB * NE) return;
  int e = i & 511;
  int b = (i >> 9) & 63;
  int t = i >> 15;                              // 64*512 = 2^15
  int c = caps[b * NT + t];
  dst[i] = (__bf16)table[(size_t)c * NE + e];
}

__global__ void __launch_bounds__(256)
cvt_f32_bf16_kernel(const float* __restrict__ s, __bf16* __restrict__ d, int n) {
  int i = blockIdx.x * 256 + threadIdx.x, st = gridDim.x * 256;
  for (; i < n; i += st) d[i] = (__bf16)s[i];
}

__global__ void __launch_bounds__(256)
zero_f32_kernel(float* __restrict__ p, int n) {
  int i = blockIdx.x * 256 + threadIdx.x, st = gridDim.x * 256;
  for (; i < n; i += st) p[i] = 0.0f;
}

__global__ void __launch_bounds__(256)
zero_bf16_kernel(__bf16* __restrict__ p, int n) {
  int i = blockIdx.x * 256 + threadIdx.x, st = gridDim.x * 256;
  for (; i < n; i += st) p[i] = (__bf16)0.0f;
}

// ---------------------------------------------------------------------------
// Generic GEMM, 64x32 output per wave (4 M-tiles x 2 N-tiles):
//   C(MxN,f32) = A(MxK,bf16,lda) @ W(NxK,bf16,ldw, col offset koff)^T + biases
// M must be a multiple of 64; N a multiple of 16 (odd half-block guarded).
// Per K-step: 4 A-frags + 2 B-frags feed 8 WMMAs (2.7x less operand traffic
// than wave-per-tile).
// ---------------------------------------------------------------------------
__global__ void __launch_bounds__(256)
gemm_bf16_kernel(const __bf16* __restrict__ A, int lda,
                 const __bf16* __restrict__ W, int ldw, int koff,
                 const float* __restrict__ bias0, const float* __restrict__ bias1,
                 float* __restrict__ C, int ldc,
                 int Mblk, int Nblk2, int N, int K) {
  int gw = (blockIdx.x * 256 + threadIdx.x) >> 5;
  if (gw >= Mblk * Nblk2) return;              // wave-uniform exit
  int bm = gw / Nblk2, bn = gw - bm * Nblk2;
  int lane = threadIdx.x & 31;
  int hi = lane >> 4, lq = lane & 15;
  int row0 = bm * 64, n0 = bn * 32;
  bool n1ok = (n0 + 16) < N;                   // wave-uniform
  int n0b = n1ok ? n0 + 16 : n0;
  v8f acc[4][2] = {};
  for (int k0 = 0; k0 < K; k0 += 32) {
    v16bf b0 = load_b_frag(W, ldw, n0,  koff + k0, lane);
    v16bf b1 = load_b_frag(W, ldw, n0b, koff + k0, lane);
#pragma unroll
    for (int mi = 0; mi < 4; mi++) {
      v16bf a = load_a_frag(A, lda, row0 + mi * 16, k0, lane);
      acc[mi][0] = wmma_bf16(a, b0, acc[mi][0]);
      acc[mi][1] = wmma_bf16(a, b1, acc[mi][1]);
    }
  }
  float bv0 = 0.0f, bv1 = 0.0f;
  if (bias0) { bv0 += bias0[n0 + lq]; bv1 += bias0[n0b + lq]; }
  if (bias1) { bv0 += bias1[n0 + lq]; bv1 += bias1[n0b + lq]; }
#pragma unroll
  for (int mi = 0; mi < 4; mi++) {
#pragma unroll
    for (int i = 0; i < 8; i++) {
      int r = row0 + mi * 16 + i + 8 * hi;
      C[(size_t)r * ldc + n0 + lq] = acc[mi][0][i] + bv0;
      if (n1ok) C[(size_t)r * ldc + n0b + lq] = acc[mi][1][i] + bv1;
    }
  }
}

// ---------------------------------------------------------------------------
// Fused LSTM cell step:
//   gates(64x2048) = base + A1(64x512)@W1^T [+ A2(64x512)@W2^T]
//   i,f,g,o split -> c_new, h_new; mask-blend carries; emit raw h (bf16).
// Grid: 16 blocks (hidden slice j of 32 units), 256 threads = 8 waves.
// Wave w owns 16 gate columns over all 64 rows; B-fragment loaded once per
// K-step and reused across the 4 M-tiles (4 live accumulators) to shorten
// the latency-critical sequential chain.
// ---------------------------------------------------------------------------
template <bool TWO_A>
__global__ void __launch_bounds__(256)
lstm_cell_kernel(const float* __restrict__ base,            // [64][2048] incl. biases
                 const __bf16* __restrict__ A1, const __bf16* __restrict__ W1, int ldw1,
                 const __bf16* __restrict__ A2, const __bf16* __restrict__ W2,
                 const float* __restrict__ c_prev, const float* __restrict__ h_prev,
                 const int* __restrict__ lens, int t,
                 float* __restrict__ h_carry, __bf16* __restrict__ h_carry_bf,
                 float* __restrict__ c_carry,
                 __bf16* __restrict__ h_raw, int raw_stride) {
  __shared__ float gsh[64 * 128];
  const int j    = blockIdx.x;          // 0..15
  const int tid  = threadIdx.x;
  const int wave = tid >> 5;            // 0..7
  const int lane = tid & 31;
  const int hi   = lane >> 4, lq = lane & 15;
  const int gate  = wave >> 1;
  const int ncol0 = gate * NH + j * 32 + (wave & 1) * 16;   // global gate column base

  v8f acc[4];
#pragma unroll
  for (int mt = 0; mt < 4; ++mt)
#pragma unroll
    for (int i = 0; i < 8; i++)
      acc[mt][i] = base[(size_t)(mt * 16 + i + 8 * hi) * G4H + ncol0 + lq];

  for (int k0 = 0; k0 < NH; k0 += 32) {
    v16bf b = load_b_frag(W1, ldw1, ncol0, k0, lane);
#pragma unroll
    for (int mt = 0; mt < 4; ++mt) {
      v16bf a = load_a_frag(A1, NH, mt * 16, k0, lane);
      acc[mt] = wmma_bf16(a, b, acc[mt]);
    }
  }
  if (TWO_A) {
    for (int k0 = 0; k0 < NH; k0 += 32) {
      v16bf b = load_b_frag(W2, NH, ncol0, k0, lane);
#pragma unroll
      for (int mt = 0; mt < 4; ++mt) {
        v16bf a = load_a_frag(A2, NH, mt * 16, k0, lane);
        acc[mt] = wmma_bf16(a, b, acc[mt]);
      }
    }
  }
#pragma unroll
  for (int mt = 0; mt < 4; ++mt)
#pragma unroll
    for (int i = 0; i < 8; i++)
      gsh[(mt * 16 + i + 8 * hi) * 128 + wave * 16 + lq] = acc[mt][i];
  __syncthreads();

  // element-wise LSTM update over this block's 64x32 hidden units
#pragma unroll
  for (int r = 0; r < 8; r++) {
    int e  = r * 256 + tid;              // 0..2047
    int b_ = e >> 5, nn = e & 31;
    int n  = j * 32 + nn;
    float ig = gsh[b_ * 128 +  0 + nn];
    float fg = gsh[b_ * 128 + 32 + nn];
    float gg = gsh[b_ * 128 + 64 + nn];
    float og = gsh[b_ * 128 + 96 + nn];
    float cp = c_prev[b_ * NH + n];
    float hp = h_prev[b_ * NH + n];
    float cn = sigmoidf_(fg) * cp + sigmoidf_(ig) * tanhf(gg);
    float hn = sigmoidf_(og) * tanhf(cn);
    bool  m  = t < (lens[b_] - 1);
    float hc = m ? hn : hp;
    float cc = m ? cn : cp;
    h_carry[b_ * NH + n]    = hc;
    c_carry[b_ * NH + n]    = cc;
    h_carry_bf[b_ * NH + n] = (__bf16)hc;
    h_raw[(size_t)b_ * raw_stride + n] = (__bf16)hn;
  }
}

// ---------------------------------------------------------------------------
// Final FC: preds(3264 x 10000) = H1(3264x512 bf16) @ fc_w^T + fc_b, masked.
// 64x32 per wave; N=10000 has a trailing 16-wide half-block (guarded).
// Row r = b*51 + t; invalid (t >= lens[b]-1) rows -> 0.
// ---------------------------------------------------------------------------
__global__ void __launch_bounds__(256)
fc_masked_kernel(const __bf16* __restrict__ A, const __bf16* __restrict__ W,
                 const float* __restrict__ bias, const int* __restrict__ lens,
                 float* __restrict__ out) {
  const int Mblk  = (NB * TD) / 64;    // 51
  const int Nblk2 = (NV / 16 + 1) / 2; // 313 (last block is a half block)
  int gw = (blockIdx.x * 256 + threadIdx.x) >> 5;
  if (gw >= Mblk * Nblk2) return;
  int bm = gw / Nblk2, bn = gw - bm * Nblk2;
  int lane = threadIdx.x & 31;
  int hi = lane >> 4, lq = lane & 15;
  int row0 = bm * 64, n0 = bn * 32;
  bool n1ok = (n0 + 16) < NV;
  int n0b = n1ok ? n0 + 16 : n0;
  v8f acc[4][2] = {};
  for (int k0 = 0; k0 < NH; k0 += 32) {
    v16bf b0 = load_b_frag(W, NH, n0,  k0, lane);
    v16bf b1 = load_b_frag(W, NH, n0b, k0, lane);
#pragma unroll
    for (int mi = 0; mi < 4; mi++) {
      v16bf a = load_a_frag(A, NH, row0 + mi * 16, k0, lane);
      acc[mi][0] = wmma_bf16(a, b0, acc[mi][0]);
      acc[mi][1] = wmma_bf16(a, b1, acc[mi][1]);
    }
  }
  float bv0 = bias[n0 + lq], bv1 = bias[n0b + lq];
#pragma unroll
  for (int mi = 0; mi < 4; mi++) {
#pragma unroll
    for (int i = 0; i < 8; i++) {
      int r  = row0 + mi * 16 + i + 8 * hi;
      int b_ = r / TD, tt = r - b_ * TD;
      bool m = tt < (lens[b_] - 1);
      out[(size_t)r * NV + n0 + lq] = m ? (acc[mi][0][i] + bv0) : 0.0f;
      if (n1ok) out[(size_t)r * NV + n0b + lq] = m ? (acc[mi][1][i] + bv1) : 0.0f;
    }
  }
}

// ---------------------------------------------------------------------------
// Host orchestration
// ---------------------------------------------------------------------------
static inline int wave_blocks(int waves) { return (waves + 7) / 8; }

extern "C" void kernel_launch(void* const* d_in, const int* in_sizes, int n_in,
                              void* d_out, int out_size, void* d_ws, size_t ws_size,
                              hipStream_t stream) {
  (void)in_sizes; (void)n_in; (void)out_size; (void)ws_size;

  const float* enc      = (const float*)d_in[0];
  const int*   caps_in  = (const int*)d_in[1];
  const int*   caplen   = (const int*)d_in[2];
  const float* emb_tab[2] = { (const float*)d_in[3], (const float*)d_in[4] };
  const float* w_ih1[2] = { (const float*)d_in[5],  (const float*)d_in[13] };
  const float* w_hh1[2] = { (const float*)d_in[6],  (const float*)d_in[14] };
  const float* b_ih1[2] = { (const float*)d_in[7],  (const float*)d_in[15] };
  const float* b_hh1[2] = { (const float*)d_in[8],  (const float*)d_in[16] };
  const float* w_ih2[2] = { (const float*)d_in[9],  (const float*)d_in[17] };
  const float* w_hh2[2] = { (const float*)d_in[10], (const float*)d_in[18] };
  const float* b_ih2[2] = { (const float*)d_in[11], (const float*)d_in[19] };
  const float* b_hh2[2] = { (const float*)d_in[12], (const float*)d_in[20] };
  const float* img_w[2] = { (const float*)d_in[21], (const float*)d_in[23] };
  const float* img_b[2] = { (const float*)d_in[22], (const float*)d_in[24] };
  const float* fc_w[2]  = { (const float*)d_in[25], (const float*)d_in[27] };
  const float* fc_b[2]  = { (const float*)d_in[26], (const float*)d_in[28] };

  float* out = (float*)d_out;
  const size_t PRED = (size_t)NB * TD * NV;           // 32,640,000
  float* out_caps     = out + 2 * PRED;
  float* out_caps_rev = out_caps + NB * NT;
  float* out_lens     = out_caps_rev + NB * NT;
  float* out_order    = out_lens + NB;

  // ---- workspace layout (bump allocator, 256B aligned) ----
  uintptr_t cur = (uintptr_t)d_ws;
  auto alloc = [&](size_t bytes) -> void* {
    cur = (cur + 255) & ~(uintptr_t)255;
    void* r = (void*)cur; cur += bytes; return r;
  };
  int* lens_s   = (int*)alloc(NB * 4);
  int* order_i  = (int*)alloc(NB * 4);
  int* caps_s   = (int*)alloc(NB * NT * 4);
  int* caps_rev = (int*)alloc(NB * NT * 4);
  __bf16* enc_s = (__bf16*)alloc((size_t)NB * NDE * 2);

  struct Dir {
    __bf16 *Wih1, *Whh1, *Wih2, *Whh2, *Wimg, *Wfc;
    __bf16 *Xemb, *imgbf, *hbf0, *hbf1, *h1bf0, *h1bf1, *hnraw, *H1;
    float  *Xpre, *imgf, *Gimg, *h, *c, *h1, *c1;
  } D[2];
  for (int d = 0; d < 2; d++) {
    D[d].Wih1 = (__bf16*)alloc((size_t)G4H * NE * 2);
    D[d].Whh1 = (__bf16*)alloc((size_t)G4H * NH * 2);
    D[d].Wih2 = (__bf16*)alloc((size_t)G4H * (2 * NH) * 2);
    D[d].Whh2 = (__bf16*)alloc((size_t)G4H * NH * 2);
    D[d].Wimg = (__bf16*)alloc((size_t)NH * NDE * 2);
    D[d].Wfc  = (__bf16*)alloc((size_t)NV * NH * 2);
    D[d].Xemb = (__bf16*)alloc((size_t)TD * NB * NE * 2);
    D[d].Xpre = (float*)alloc((size_t)TD * NB * G4H * 4);
    D[d].imgf = (float*)alloc((size_t)NB * NH * 4);
    D[d].imgbf= (__bf16*)alloc((size_t)NB * NH * 2);
    D[d].Gimg = (float*)alloc((size_t)NB * G4H * 4);
    D[d].h    = (float*)alloc((size_t)NB * NH * 4);
    D[d].c    = (float*)alloc((size_t)NB * NH * 4);
    D[d].h1   = (float*)alloc((size_t)NB * NH * 4);
    D[d].c1   = (float*)alloc((size_t)NB * NH * 4);
    D[d].hbf0 = (__bf16*)alloc((size_t)NB * NH * 2);
    D[d].hbf1 = (__bf16*)alloc((size_t)NB * NH * 2);
    D[d].h1bf0= (__bf16*)alloc((size_t)NB * NH * 2);
    D[d].h1bf1= (__bf16*)alloc((size_t)NB * NH * 2);
    D[d].hnraw= (__bf16*)alloc((size_t)NB * NH * 2);
    D[d].H1   = (__bf16*)alloc((size_t)NB * TD * NH * 2);
  }

  // ---- prep: sort, reverse, int outputs ----
  prep_kernel<<<1, 64, 0, stream>>>(caplen, caps_in, lens_s, order_i, caps_s,
                                    caps_rev, out_caps, out_caps_rev, out_lens,
                                    out_order);
  enc_gather_kernel<<<(NB * NDE + 255) / 256, 256, 0, stream>>>(enc, order_i, enc_s);

  for (int d = 0; d < 2; d++) {
    Dir& W = D[d];
    auto cvt = [&](const float* s, __bf16* dst, int n) {
      int blocks = (n + 1023) / 1024; if (blocks > 4096) blocks = 4096;
      cvt_f32_bf16_kernel<<<blocks, 256, 0, stream>>>(s, dst, n);
    };
    cvt(w_ih1[d], W.Wih1, G4H * NE);
    cvt(w_hh1[d], W.Whh1, G4H * NH);
    cvt(w_ih2[d], W.Wih2, G4H * 2 * NH);
    cvt(w_hh2[d], W.Whh2, G4H * NH);
    cvt(img_w[d], W.Wimg, NH * NDE);
    cvt(fc_w[d],  W.Wfc,  NV * NH);

    // embeddings (sorted forward / reversed), [t][b][E] bf16
    const int* ctab = (d == 0) ? caps_s : caps_rev;
    embed_kernel<<<(TD * NB * NE + 255) / 256, 256, 0, stream>>>(emb_tab[d], ctab,
                                                                 W.Xemb);

    // Xpre = Xemb @ w_ih1^T + b_ih1 + b_hh1   (3264 x 2048, K=512)
    {
      int mb = NB * TD / 64, nb2 = G4H / 32;
      gemm_bf16_kernel<<<wave_blocks(mb * nb2), 256, 0, stream>>>(
          W.Xemb, NE, W.Wih1, NE, 0, b_ih1[d], b_hh1[d], W.Xpre, G4H,
          mb, nb2, G4H, NE);
    }
    // img features: enc_s @ img_w^T + img_b    (64 x 512, K=2048)
    {
      int mb = NB / 64, nb2 = NH / 32;
      gemm_bf16_kernel<<<wave_blocks(mb * nb2), 256, 0, stream>>>(
          enc_s, NDE, W.Wimg, NDE, 0, img_b[d], nullptr, W.imgf, NH,
          mb, nb2, NH, NDE);
    }
    cvt(W.imgf, W.imgbf, NB * NH);

    // Gimg = img @ w_ih2[:,512:]^T + b_ih2 + b_hh2   (64 x 2048, K=512)
    {
      int mb = NB / 64, nb2 = G4H / 32;
      gemm_bf16_kernel<<<wave_blocks(mb * nb2), 256, 0, stream>>>(
          W.imgbf, NH, W.Wih2, 2 * NH, NH, b_ih2[d], b_hh2[d], W.Gimg, G4H,
          mb, nb2, G4H, NH);
    }

    // zero initial states
    zero_f32_kernel<<<32, 256, 0, stream>>>(W.h,  NB * NH);
    zero_f32_kernel<<<32, 256, 0, stream>>>(W.c,  NB * NH);
    zero_f32_kernel<<<32, 256, 0, stream>>>(W.h1, NB * NH);
    zero_f32_kernel<<<32, 256, 0, stream>>>(W.c1, NB * NH);
    zero_bf16_kernel<<<32, 256, 0, stream>>>(W.hbf0,  NB * NH);
    zero_bf16_kernel<<<32, 256, 0, stream>>>(W.h1bf0, NB * NH);

    // sequential recurrence: 51 steps, two fused cell kernels per step
    for (int t = 0; t < TD; t++) {
      __bf16* hcur  = (t & 1) ? W.hbf1  : W.hbf0;
      __bf16* hnxt  = (t & 1) ? W.hbf0  : W.hbf1;
      __bf16* h1cur = (t & 1) ? W.h1bf1 : W.h1bf0;
      __bf16* h1nxt = (t & 1) ? W.h1bf0 : W.h1bf1;

      // cell 1: gates = Xpre[t] + h @ w_hh1^T
      lstm_cell_kernel<false><<<16, 256, 0, stream>>>(
          W.Xpre + (size_t)t * NB * G4H,
          hcur, W.Whh1, NH, nullptr, nullptr,
          W.c, W.h, lens_s, t,
          W.h, hnxt, W.c, W.hnraw, NH);

      // cell 2: gates = Gimg + hn_raw @ w_ih2[:,:512]^T + h1 @ w_hh2^T
      lstm_cell_kernel<true><<<16, 256, 0, stream>>>(
          W.Gimg,
          W.hnraw, W.Wih2, 2 * NH, h1cur, W.Whh2,
          W.c1, W.h1, lens_s, t,
          W.h1, h1nxt, W.c1, W.H1 + (size_t)t * NH, TD * NH);
    }

    // masked FC: preds[d] = H1 @ fc_w^T + fc_b  (3264 x 10000, K=512)
    {
      int mb = (NB * TD) / 64, nb2 = (NV / 16 + 1) / 2;
      fc_masked_kernel<<<wave_blocks(mb * nb2), 256, 0, stream>>>(
          W.H1, W.Wfc, fc_b[d], lens_s, out + (size_t)d * PRED);
    }
  }
}
